// Clos_34282428956960
// MI455X (gfx1250) — matrix operand order repaired and checked by reading, not compiled, not run on hardware
//
#include <hip/hip_runtime.h>
#include <hip/hip_bf16.h>

typedef __attribute__((ext_vector_type(16))) _Float16 v16h;
typedef __attribute__((ext_vector_type(2)))  _Float16 v2h;
typedef __attribute__((ext_vector_type(8)))  float    v8f;

#define BC_DIM   8192   // B*C rows
#define D_IN     4096
#define D_OUT    4096
#define MID      256    // B2

// workspace layout (bytes), all 256B aligned
#define OFF_W2S   0u                         // 64*256 f32   = 64 KB
#define OFF_CBIAS (OFF_W2S + 64u*256u*4u)    // 256 f32
#define OFF_A     (OFF_CBIAS + 4096u)        // 4096*256 f16 = 2 MB
#define OFF_B     (OFF_A + 4096u*256u*2u)    // 256*4096 f16 = 2 MB
#define OFF_G     (OFF_B + 256u*4096u*2u)    // 8192*256 f16 = 4 MB

// ---------------- prep kernels ----------------

// w2s[r*256+m] = sum_n w2[r][m][n]   (w2 flat: ((r*256+m)*64)+n)
__global__ __launch_bounds__(256) void clos_prep_w2s(const float* __restrict__ w2,
                                                     float* __restrict__ w2s) {
  const int tid = blockIdx.x * 256 + threadIdx.x;   // 16384 threads
  const float* p = w2 + (size_t)tid * 64;
  float s = 0.f;
#pragma unroll
  for (int n = 0; n < 64; ++n) s += p[n];
  w2s[tid] = s;
}

// A[d*256+m] = f16( w1[n][r][m] * w2s[r][m] ),  d = n*64+r
__global__ __launch_bounds__(256) void clos_prep_A(const float* __restrict__ w1,
                                                   const float* __restrict__ w2s,
                                                   _Float16* __restrict__ A) {
  const int tid = blockIdx.x * 256 + threadIdx.x;   // 1048576 threads
  const int d = tid >> 8, m = tid & 255;
  const int n = d >> 6, r = d & 63;
  const float v = w1[((size_t)(n * 64 + r)) * 256 + m] * w2s[r * 256 + m];
  A[tid] = (_Float16)v;
}

// cbias[m] = sum_r bias1[r]*w2s[r][m] + 64*bias2[m]
__global__ __launch_bounds__(256) void clos_prep_cbias(const float* __restrict__ bias1,
                                                       const float* __restrict__ bias2,
                                                       const float* __restrict__ w2s,
                                                       float* __restrict__ cbias) {
  const int m = threadIdx.x;
  float s = 0.f;
#pragma unroll
  for (int r = 0; r < 64; ++r) s += bias1[r] * w2s[r * 256 + m];
  cbias[m] = s + 64.0f * bias2[m];
}

// Bm[m*4096 + j] = f16( w3[m][r][o] ), j = o*64 + r  (w3 flat: ((m*64+r)*64)+o)
__global__ __launch_bounds__(256) void clos_prep_B(const float* __restrict__ w3,
                                                   _Float16* __restrict__ Bm) {
  const int tid = blockIdx.x * 256 + threadIdx.x;   // 1048576 threads
  const int m = tid >> 12, j = tid & 4095;
  const int r = j & 63, o = j >> 6;
  Bm[tid] = (_Float16)w3[((size_t)(m * 64 + r)) * 64 + o];
}

// ---------------- GEMM 1: G[8192,256] = f16( X[8192,4096] @ A[4096,256] + cbias ) ----------------

__global__ __launch_bounds__(256) void clos_gemm1(const float* __restrict__ X,
                                                  const _Float16* __restrict__ A,
                                                  const float* __restrict__ cbias,
                                                  _Float16* __restrict__ G) {
  __shared__ _Float16 XS[32 * 40];    // 32 rows x 32 k (f16), stride 40
  __shared__ _Float16 AS[32 * 264];   // 32 k x 256 n, stride 264

  const int t    = threadIdx.x;
  const int lane = t & 31;
  const int wave = t >> 5;
  const int wm   = wave >> 2;   // 0..1 : row strip of 16
  const int wn   = wave & 3;    // 0..3 : 64-wide col strip
  const int half = lane >> 4;
  const int mrow = lane & 15;
  const int rowBase = blockIdx.x * 32;

  v8f acc[4];
#pragma unroll
  for (int i = 0; i < 4; ++i)
    acc[i] = (v8f){0.f, 0.f, 0.f, 0.f, 0.f, 0.f, 0.f, 0.f};

  for (int k0 = 0; k0 < D_IN; k0 += 32) {
    // stage X tile: fp32 -> f16 into LDS (one float4 per thread)
    {
      const int r = t >> 3, q = t & 7;
      const float4 v = *(const float4*)(X + (size_t)(rowBase + r) * D_IN + k0 + q * 4);
      _Float16* p = &XS[r * 40 + q * 4];
      p[0] = (_Float16)v.x; p[1] = (_Float16)v.y;
      p[2] = (_Float16)v.z; p[3] = (_Float16)v.w;
    }
    // stage A tile [32][256] f16 via 128b loads
#pragma unroll
    for (int ii = 0; ii < 4; ++ii) {
      const int idx = t + 256 * ii;           // 0..1023 uint4 chunks
      const int kk = idx >> 5, cu = idx & 31;
      const uint4 v = *(const uint4*)(A + (size_t)(k0 + kk) * MID + cu * 8);
      *(uint4*)(&AS[kk * 264 + cu * 8]) = v;
    }
    __syncthreads();

    // A-fragment (16x32, row = lane%16, half/K interleave per ISA 7.12.2)
    v16h af;
    {
      const _Float16* xr = &XS[(wm * 16 + mrow) * 40];
#pragma unroll
      for (int i = 0; i < 8; ++i) {
        const int kk = (i < 4) ? (half * 8 + 2 * i) : (16 + half * 8 + 2 * (i - 4));
        const v2h p = *(const v2h*)(xr + kk);
        af[2 * i] = p[0]; af[2 * i + 1] = p[1];
      }
    }
#pragma unroll
    for (int nt = 0; nt < 4; ++nt) {
      v16h bf;
      const _Float16* br = &AS[wn * 64 + nt * 16 + mrow];
#pragma unroll
      for (int j = 0; j < 8; ++j) {
        const int kk = (j < 4) ? (half * 8 + 2 * j) : (16 + half * 8 + 2 * (j - 4));
        bf[2 * j] = br[kk * 264]; bf[2 * j + 1] = br[(kk + 1) * 264];
      }
      acc[nt] = __builtin_amdgcn_wmma_f32_16x16x32_f16(false, af, false, bf,
                                                       (short)0, acc[nt], false, false);
    }
    __syncthreads();
  }

  // epilogue: C/D layout — VGPR i: M = i + 8*half, N = lane%16
  const int rbase = rowBase + wm * 16 + 8 * half;
#pragma unroll
  for (int nt = 0; nt < 4; ++nt) {
    const int col = wn * 64 + nt * 16 + mrow;
    const float cb = cbias[col];
#pragma unroll
    for (int i = 0; i < 8; ++i)
      G[(size_t)(rbase + i) * MID + col] = (_Float16)(acc[nt][i] + cb);
  }
}

// ---------------- GEMM 2: OUT[8192,4096] = G[8192,256] @ B[256,4096] + bias3[col&63] ----------------

__global__ __launch_bounds__(256) void clos_gemm2(const _Float16* __restrict__ G,
                                                  const _Float16* __restrict__ Bm,
                                                  const float* __restrict__ bias3,
                                                  float* __restrict__ OUT) {
  __shared__ _Float16 GS[32 * 40];
  __shared__ _Float16 BS[32 * 264];

  const int t    = threadIdx.x;
  const int lane = t & 31;
  const int wave = t >> 5;
  const int wm   = wave >> 2;
  const int wn   = wave & 3;
  const int half = lane >> 4;
  const int mrow = lane & 15;
  const int rowBase = blockIdx.y * 32;
  const int nBase   = blockIdx.x * 256;

  v8f acc[4];
#pragma unroll
  for (int i = 0; i < 4; ++i)
    acc[i] = (v8f){0.f, 0.f, 0.f, 0.f, 0.f, 0.f, 0.f, 0.f};

  for (int k0 = 0; k0 < MID; k0 += 32) {
    // stage G tile [32 rows][32 k] f16 (128 threads, one uint4 each)
    if (t < 128) {
      const int r = t >> 2, cu = t & 3;
      const uint4 v = *(const uint4*)(G + (size_t)(rowBase + r) * MID + k0 + cu * 8);
      *(uint4*)(&GS[r * 40 + cu * 8]) = v;
    }
    // stage B tile [32 k][256 n] f16
#pragma unroll
    for (int ii = 0; ii < 4; ++ii) {
      const int idx = t + 256 * ii;
      const int kk = idx >> 5, cu = idx & 31;
      const uint4 v = *(const uint4*)(Bm + (size_t)(k0 + kk) * D_OUT + nBase + cu * 8);
      *(uint4*)(&BS[kk * 264 + cu * 8]) = v;
    }
    __syncthreads();

    v16h af;
    {
      const _Float16* gr = &GS[(wm * 16 + mrow) * 40];
#pragma unroll
      for (int i = 0; i < 8; ++i) {
        const int kk = (i < 4) ? (half * 8 + 2 * i) : (16 + half * 8 + 2 * (i - 4));
        const v2h p = *(const v2h*)(gr + kk);
        af[2 * i] = p[0]; af[2 * i + 1] = p[1];
      }
    }
#pragma unroll
    for (int nt = 0; nt < 4; ++nt) {
      v16h bf;
      const _Float16* br = &BS[wn * 64 + nt * 16 + mrow];
#pragma unroll
      for (int j = 0; j < 8; ++j) {
        const int kk = (j < 4) ? (half * 8 + 2 * j) : (16 + half * 8 + 2 * (j - 4));
        bf[2 * j] = br[kk * 264]; bf[2 * j + 1] = br[(kk + 1) * 264];
      }
      acc[nt] = __builtin_amdgcn_wmma_f32_16x16x32_f16(false, af, false, bf,
                                                       (short)0, acc[nt], false, false);
    }
    __syncthreads();
  }

  const int rbase = rowBase + wm * 16 + 8 * half;
#pragma unroll
  for (int nt = 0; nt < 4; ++nt) {
    const int col = nBase + wn * 64 + nt * 16 + mrow;
    const float b3 = bias3[col & 63];
#pragma unroll
    for (int i = 0; i < 8; ++i)
      OUT[(size_t)(rbase + i) * D_OUT + col] = acc[nt][i] + b3;
  }
}

// ---------------- launch ----------------

extern "C" void kernel_launch(void* const* d_in, const int* in_sizes, int n_in,
                              void* d_out, int out_size, void* d_ws, size_t ws_size,
                              hipStream_t stream) {
  (void)in_sizes; (void)n_in; (void)out_size; (void)ws_size;
  const float* x  = (const float*)d_in[0];
  const float* w1 = (const float*)d_in[1];
  const float* w2 = (const float*)d_in[2];
  const float* w3 = (const float*)d_in[3];
  const float* b1 = (const float*)d_in[4];
  const float* b2 = (const float*)d_in[5];
  const float* b3 = (const float*)d_in[6];

  char* ws = (char*)d_ws;
  float*    w2s   = (float*)(ws + OFF_W2S);
  float*    cbias = (float*)(ws + OFF_CBIAS);
  _Float16* A     = (_Float16*)(ws + OFF_A);
  _Float16* Bm    = (_Float16*)(ws + OFF_B);
  _Float16* G     = (_Float16*)(ws + OFF_G);
  float*    out   = (float*)d_out;

  clos_prep_w2s  <<<64,   256, 0, stream>>>(w2, w2s);
  clos_prep_A    <<<4096, 256, 0, stream>>>(w1, w2s, A);
  clos_prep_cbias<<<1,    256, 0, stream>>>(b1, b2, w2s, cbias);
  clos_prep_B    <<<4096, 256, 0, stream>>>(w3, Bm);

  clos_gemm1<<<BC_DIM / 32, 256, 0, stream>>>(x, A, cbias, G);
  clos_gemm2<<<dim3(D_OUT / 256, BC_DIM / 32), 256, 0, stream>>>(G, Bm, b3, out);
}
